// SeparationParamSimplest_42125039239354
// MI455X (gfx1250) — matrix operand order repaired and checked by reading, not compiled, not run on hardware
//
#include <hip/hip_runtime.h>
#include <hip/hip_bf16.h>

typedef __attribute__((ext_vector_type(16))) _Float16 v16h;
typedef __attribute__((ext_vector_type(8)))  float    v8f;

#define DIM 32
#define NB  6
#define NN  256
#define BB  16

// workspace layout (bytes)
#define CFB_OFF   0u                            // [BB][NB][DIM] f32 = 12288
#define CGB_OFF   (BB * NB * DIM * 4u)          // 12288..24575
#define AFRAG_OFF (2u * BB * NB * DIM * 4u)     // 24576..: 36 frags * 32 lanes * 32B = 36864

// ---------------------------------------------------------------------------
// Prep A: per-batch constants.
//   ex = sin(xp@xW1+xb1)@xW2+xb2 ; et likewise.
//   cfb[b,k,:] = et[b]@f_Wz[k][2:34] + f_bz[k] + f_bh[k]
//   cgb[b,k,:] = ex[b]@g_Wz[k][32:64] + g_bz[k] + g_bh[k]
// ---------------------------------------------------------------------------
__global__ __launch_bounds__(192) void prep_batch_kernel(
    const float* __restrict__ x_params, const float* __restrict__ t_params,
    const float* __restrict__ xW1, const float* __restrict__ xb1,
    const float* __restrict__ xW2, const float* __restrict__ xb2,
    const float* __restrict__ tW1, const float* __restrict__ tb1,
    const float* __restrict__ tW2, const float* __restrict__ tb2,
    const float* __restrict__ fWz, const float* __restrict__ fbz, const float* __restrict__ fbh,
    const float* __restrict__ gWz, const float* __restrict__ gbz, const float* __restrict__ gbh,
    float* __restrict__ cfb, float* __restrict__ cgb)
{
    const int b   = blockIdx.x;
    const int tid = threadIdx.x;
    __shared__ float hx[DIM], ht[DIM], ex[DIM], et[DIM];

    if (tid < DIM) {
        float a = xb1[tid];
        for (int c = 0; c < 16; ++c) a += x_params[b * 16 + c] * xW1[c * DIM + tid];
        hx[tid] = __sinf(a);
    } else if (tid < 2 * DIM) {
        const int d = tid - DIM;
        float a = tb1[d];
        for (int c = 0; c < 8; ++c) a += t_params[b * 8 + c] * tW1[c * DIM + d];
        ht[d] = __sinf(a);
    }
    __syncthreads();
    if (tid < DIM) {
        float a = xb2[tid];
        for (int c = 0; c < DIM; ++c) a += hx[c] * xW2[c * DIM + tid];
        ex[tid] = a;
    } else if (tid < 2 * DIM) {
        const int d = tid - DIM;
        float a = tb2[d];
        for (int c = 0; c < DIM; ++c) a += ht[c] * tW2[c * DIM + d];
        et[d] = a;
    }
    __syncthreads();

    const int k = tid >> 5;      // 0..5
    const int d = tid & 31;
    float af = fbz[k * DIM + d] + fbh[k * DIM + d];
    for (int c = 0; c < DIM; ++c) af += et[c] * fWz[(k * 34 + 2 + c) * DIM + d];
    cfb[(b * NB + k) * DIM + d] = af;

    float ag = gbz[k * DIM + d] + gbh[k * DIM + d];
    for (int c = 0; c < DIM; ++c) ag += ex[c] * gWz[(k * 64 + 32 + c) * DIM + d];
    cgb[(b * NB + k) * DIM + d] = ag;
}

// ---------------------------------------------------------------------------
// Prep B: pack W^T halves into WMMA A fragments (f16, 16x32, K=32) with a
// PERMUTED output-feature mapping chosen so that the WMMA D fragment of a
// lane contains exactly the 16 logical features that the same lane's next
// B fragment needs (no cross-lane shuffle between layers):
//   physical A row M of half p  ->  logical out-feature
//       o(p,M) = (M < 8) ? 8p + M : 16 + 8p + (M - 8)
// A's K index is the identity over logical in-features (state physical K ==
// logical feature).  A[M][K] = W[K][o(p,M)].
// A-layout (16-bit 16x32): lane gives M=l16; half t -> K=16*hi+t
// (contiguous, matching the B-fragment state layout).
// Matrices m: 0..5 f_Wh, 6..11 g_Wh, 12..17 g_Wz rows 0..31.
// ---------------------------------------------------------------------------
__global__ __launch_bounds__(32) void prep_frag_kernel(
    const float* __restrict__ fWh, const float* __restrict__ gWh,
    const float* __restrict__ gWz, v16h* __restrict__ afrag)
{
    const int fid = blockIdx.x;         // 0..35
    const int m   = fid >> 1;
    const int p   = fid & 1;
    const int l   = threadIdx.x;
    const int l16 = l & 15;
    const int hi  = l >> 4;

    const float* W;
    if (m < 6)       W = fWh + m * 1024;
    else if (m < 12) W = gWh + (m - 6) * 1024;
    else             W = gWz + (m - 12) * 2048;  // rows 0..31 of [64][32]

    const int col = (l16 < 8) ? (8 * p + l16) : (16 + 8 * p + (l16 - 8)); // o(p,M=l16)

    v16h h;
    #pragma unroll
    for (int t = 0; t < 16; ++t) {
        const int K = 16 * hi + t;
        h[t] = (_Float16)W[K * DIM + col];
    }
    afrag[fid * 32 + l] = h;
}

// ---------------------------------------------------------------------------
// Main: one block per (b,i). 8 waves, each wave runs 2 tiles of 16 j's
// through all 12 layers with V_WMMA_F32_16X16X32_F16.
// State in B-fragment layout: halfs[t] = H[pos=l16][feat=16*hi+t].
// With the permuted A packing, acc0[v] == feat 16*hi+v and
// acc1[v] == feat 16*hi+8+v for this lane, so D -> next B is a pure
// per-lane f16 convert (no shuffles, no LDS).
// Loops are rolled (#pragma unroll 1) to keep VGPR pressure low (no spills).
// ---------------------------------------------------------------------------
__global__ __launch_bounds__(256) void pinn_main_kernel(
    const float* __restrict__ coords,
    const float* __restrict__ h0,  const float* __restrict__ gh0,
    const float* __restrict__ fWz,
    const float* __restrict__ dW,  const float* __restrict__ db,
    const float* __restrict__ cfb, const float* __restrict__ cgb,
    const v16h*  __restrict__ afrag,
    float* __restrict__ out)
{
    const int bi = blockIdx.x;
    const int b  = bi >> 8;              // N = 256
    const int i  = bi & (NN - 1);

    // warm the 36KB fragment table into WGP$/L2 (global_prefetch_b8)
    __builtin_prefetch((const char*)afrag + threadIdx.x * 144, 0, 0);

    const int lane = threadIdx.x & 31;
    const int w    = threadIdx.x >> 5;
    const int l16  = lane & 15;
    const int hi   = lane >> 4;
    const int flo  = 16 * hi;            // logical feats of acc0: flo..flo+7
    const int fhi  = 16 * hi + 8;        // logical feats of acc1

    const float ti  = coords[(b * NN + i) * 2 + 1];
    const float db0 = db[0];

    #pragma unroll 1
    for (int tt = 0; tt < 2; ++tt) {
        const int   j  = (w + 8 * tt) * 16 + l16;
        const float xj = coords[(b * NN + j) * 2 + 0];

        v16h H;
        #pragma unroll
        for (int t = 0; t < 16; ++t) H[t] = (_Float16)h0[16 * hi + t];

        // -------------------- f blocks --------------------
        #pragma unroll 1
        for (int k = 0; k < NB; ++k) {
            const float* wz0 = fWz + (k * 34 + 0) * DIM;
            const float* wz1 = fWz + (k * 34 + 1) * DIM;
            const float* cf  = cfb + (b * NB + k) * DIM;
            float w0a[8], w0b[8];
            v8f c0, c1;
            #pragma unroll
            for (int v = 0; v < 8; ++v) {
                w0a[v] = wz0[flo + v];
                w0b[v] = wz0[fhi + v];
                c0[v]  = ti * wz1[flo + v] + cf[flo + v];
                c1[v]  = ti * wz1[fhi + v] + cf[fhi + v];
            }
            const v16h a0 = afrag[(2 * k + 0) * 32 + lane];
            const v16h a1 = afrag[(2 * k + 1) * 32 + lane];
            v8f acc0 = __builtin_amdgcn_wmma_f32_16x16x32_f16(false, a0, false, H, (short)0, c0, false, false);
            v8f acc1 = __builtin_amdgcn_wmma_f32_16x16x32_f16(false, a1, false, H, (short)0, c1, false, false);

            v16h Hn;
            #pragma unroll
            for (int v = 0; v < 8; ++v) {
                Hn[v]     = (_Float16)__sinf(acc0[v] + xj * w0a[v]);  // feats 16hi+0..7
                Hn[8 + v] = (_Float16)__sinf(acc1[v] + xj * w0b[v]);  // feats 16hi+8..15
            }
            H = Hn;
        }
        const v16h Hf = H;   // final f output, feeds every g-block cond

        // -------------------- g blocks --------------------
        v16h G;
        #pragma unroll
        for (int t = 0; t < 16; ++t) G[t] = (_Float16)gh0[16 * hi + t];

        #pragma unroll 1
        for (int k = 0; k < NB; ++k) {
            const float* cg = cgb + (b * NB + k) * DIM;
            v8f c0, c1;
            #pragma unroll
            for (int v = 0; v < 8; ++v) { c0[v] = cg[flo + v]; c1[v] = cg[fhi + v]; }

            const v16h awh0 = afrag[(2 * (6 + k) + 0) * 32 + lane];
            const v16h awh1 = afrag[(2 * (6 + k) + 1) * 32 + lane];
            const v16h awz0 = afrag[(2 * (12 + k) + 0) * 32 + lane];
            const v16h awz1 = afrag[(2 * (12 + k) + 1) * 32 + lane];

            v8f acc0 = __builtin_amdgcn_wmma_f32_16x16x32_f16(false, awh0, false, G,  (short)0, c0,   false, false);
            acc0     = __builtin_amdgcn_wmma_f32_16x16x32_f16(false, awz0, false, Hf, (short)0, acc0, false, false);
            v8f acc1 = __builtin_amdgcn_wmma_f32_16x16x32_f16(false, awh1, false, G,  (short)0, c1,   false, false);
            acc1     = __builtin_amdgcn_wmma_f32_16x16x32_f16(false, awz1, false, Hf, (short)0, acc1, false, false);

            v16h Gn;
            #pragma unroll
            for (int v = 0; v < 8; ++v) {
                Gn[v]     = (_Float16)__sinf(acc0[v]);
                Gn[8 + v] = (_Float16)__sinf(acc1[v]);
            }
            G = Gn;
        }

        // -------------------- decode --------------------
        float part = 0.f;
        #pragma unroll
        for (int t = 0; t < 16; ++t) part += (float)G[t] * dW[16 * hi + t];
        const float tot = part + __shfl_xor(part, 16, 32);
        if (hi == 0) out[(b * NN + i) * NN + j] = tot + db0;
    }
}

// ---------------------------------------------------------------------------
extern "C" void kernel_launch(void* const* d_in, const int* in_sizes, int n_in,
                              void* d_out, int out_size, void* d_ws, size_t ws_size,
                              hipStream_t stream)
{
    (void)in_sizes; (void)n_in; (void)out_size; (void)ws_size;
    const float* coords   = (const float*)d_in[0];
    const float* x_params = (const float*)d_in[1];
    const float* t_params = (const float*)d_in[2];
    const float* xW1      = (const float*)d_in[3];
    const float* xb1      = (const float*)d_in[4];
    const float* xW2      = (const float*)d_in[5];
    const float* xb2      = (const float*)d_in[6];
    const float* tW1      = (const float*)d_in[7];
    const float* tb1      = (const float*)d_in[8];
    const float* tW2      = (const float*)d_in[9];
    const float* tb2      = (const float*)d_in[10];
    const float* h0       = (const float*)d_in[11];
    const float* gh0      = (const float*)d_in[12];
    const float* fWh      = (const float*)d_in[13];
    const float* fbh      = (const float*)d_in[14];
    const float* fWz      = (const float*)d_in[15];
    const float* fbz      = (const float*)d_in[16];
    const float* gWh      = (const float*)d_in[17];
    const float* gbh      = (const float*)d_in[18];
    const float* gWz      = (const float*)d_in[19];
    const float* gbz      = (const float*)d_in[20];
    const float* dW       = (const float*)d_in[21];
    const float* db       = (const float*)d_in[22];

    float* cfb   = (float*)((char*)d_ws + CFB_OFF);
    float* cgb   = (float*)((char*)d_ws + CGB_OFF);
    v16h*  afrag = (v16h*)((char*)d_ws + AFRAG_OFF);

    prep_batch_kernel<<<BB, 192, 0, stream>>>(
        x_params, t_params, xW1, xb1, xW2, xb2, tW1, tb1, tW2, tb2,
        fWz, fbz, fbh, gWz, gbz, gbh, cfb, cgb);

    prep_frag_kernel<<<36, 32, 0, stream>>>(fWh, gWh, gWz, afrag);

    pinn_main_kernel<<<BB * NN, 256, 0, stream>>>(
        coords, h0, gh0, fWz, dW, db, cfb, cgb, afrag, (float*)d_out);
}
